// MedML_80994493268476
// MI455X (gfx1250) — matrix operand order repaired and checked by reading, not compiled, not run on hardware
//
#include <hip/hip_runtime.h>
#include <hip/hip_bf16.h>
#include <math.h>

// ---------------------------------------------------------------------------
// GAT forward for MI455X (gfx1250, wave32, WMMA).
// N=8192 nodes, F=128 features, 64 segments of 128 rows.
// ---------------------------------------------------------------------------

typedef __attribute__((ext_vector_type(16))) __bf16 v16bf;
typedef __attribute__((ext_vector_type(8)))  __bf16 v8bf;
typedef __attribute__((ext_vector_type(8)))  float  v8f;

#define NN    8192
#define FF    128
#define MLPD  256
#define BSEG  64
#define LSEG  128
#define KSPLIT 4
#define KCH   (NN / KSPLIT)   // 2048 columns per K-chunk wave

static __device__ __forceinline__ v8f zero8() {
  v8f z;
  #pragma unroll
  for (int i = 0; i < 8; ++i) z[i] = 0.0f;
  return z;
}

static __device__ __forceinline__ v16bf cat16(v8bf lo, v8bf hi) {
  return __builtin_shufflevector(lo, hi, 0,1,2,3,4,5,6,7,8,9,10,11,12,13,14,15);
}

static __device__ __forceinline__ float lrelu(float x) {
  return x > 0.0f ? x : 0.01f * x;
}

// ---------------------------------------------------------------------------
// K0: f32 -> bf16 conversions. g_bf row-major (WMMA A operand of g@W),
// Wt_bf = W^T row-major so each B-operand lane reads contiguous K.
// ---------------------------------------------------------------------------
__global__ void k_prep(const float* __restrict__ g, const float* __restrict__ W,
                       __bf16* __restrict__ g_bf, __bf16* __restrict__ Wt_bf) {
  int tid = blockIdx.x * blockDim.x + threadIdx.x;
  if (tid < NN * FF) g_bf[tid] = (__bf16)g[tid];
  if (tid < FF * FF) {
    int n = tid >> 7, k = tid & 127;
    Wt_bf[tid] = (__bf16)W[k * FF + n];
  }
}

// ---------------------------------------------------------------------------
// K1: h = g @ W via v_wmma_f32_16x16x32_bf16. One 16x16 tile per wave.
// Writes h (f32, for f1/f2) and hT (bf16, [col][row]) for the att@h B operand.
// A-operand per-lane K layout (16-bit, 16x32): lanes 0-15 row=lane, K=0..7 &
// 16..23; lanes 16-31 row=lane-16, K=8..15 & 24..31.
// ---------------------------------------------------------------------------
__global__ void k_gemm_h(const __bf16* __restrict__ g_bf,
                         const __bf16* __restrict__ Wt_bf,
                         float* __restrict__ h,
                         __bf16* __restrict__ hT) {
  const int wid  = blockIdx.x * (blockDim.x >> 5) + (threadIdx.x >> 5);
  const int lane = threadIdx.x & 31;
  const int rg   = wid >> 3;         // 512 row groups of 16
  const int cb   = wid & 7;          // 8 col blocks of 16
  const int l15  = lane & 15;
  const int hs   = lane >> 4;        // half-wave select
  const int koff = hs * 8;
  const int row  = rg * 16 + l15;
  const int ncol = cb * 16 + l15;

  v8f c = zero8();
  const __bf16* arow = g_bf  + (size_t)row  * FF;
  const __bf16* brow = Wt_bf + (size_t)ncol * FF;
  #pragma unroll
  for (int k = 0; k < FF; k += 32) {
    v16bf a = cat16(*(const v8bf*)(arow + k + koff),
                    *(const v8bf*)(arow + k + 16 + koff));
    v16bf b = cat16(*(const v8bf*)(brow + k + koff),
                    *(const v8bf*)(brow + k + 16 + koff));
    c = __builtin_amdgcn_wmma_f32_16x16x32_bf16(false, a, false, b,
                                                (short)0, c, false, false);
  }
  #pragma unroll
  for (int r = 0; r < 8; ++r) {
    int rr = rg * 16 + r + hs * 8;           // C layout: vgpr r -> row r (+8)
    h [(size_t)rr   * FF + ncol] = c[r];
    hT[(size_t)ncol * NN + rr  ] = (__bf16)c[r];
  }
}

// ---------------------------------------------------------------------------
// K2: f1 = h@a1, f2 = h@a2 (one thread per row).
// ---------------------------------------------------------------------------
__global__ void k_f12(const float* __restrict__ h,
                      const float* __restrict__ a1,
                      const float* __restrict__ a2,
                      float* __restrict__ f1, float* __restrict__ f2) {
  int i = blockIdx.x * blockDim.x + threadIdx.x;
  if (i >= NN) return;
  const float4* hp = (const float4*)(h + (size_t)i * FF);
  const float4* p1 = (const float4*)a1;
  const float4* p2 = (const float4*)a2;
  float s1 = 0.0f, s2 = 0.0f;
  #pragma unroll
  for (int k = 0; k < FF / 4; ++k) {
    float4 hv = hp[k], v1 = p1[k], v2 = p2[k];
    s1 += hv.x * v1.x + hv.y * v1.y + hv.z * v1.z + hv.w * v1.w;
    s2 += hv.x * v2.x + hv.y * v2.y + hv.z * v2.z + hv.w * v2.w;
  }
  f1[i] = s1;
  f2[i] = s2;
}

// ---------------------------------------------------------------------------
// K3: per-row masked-softmax stats. One wave per row; online (max,sum) per
// lane over strided int4/float4 chunks, then wave32 shfl-xor merge.
// First adj sweep (256 MB).
// ---------------------------------------------------------------------------
__global__ void k_rowstats(const int* __restrict__ adj,
                           const float* __restrict__ f1,
                           const float* __restrict__ f2,
                           float* __restrict__ rowmax,
                           float* __restrict__ rowinv) {
  const int i    = blockIdx.x * (blockDim.x >> 5) + (threadIdx.x >> 5);
  const int lane = threadIdx.x & 31;
  const float f1r = f1[i];
  const size_t rbase = (size_t)i * NN;
  float m = -1.0e30f, s = 0.0f;
  for (int t = 0; t < NN / 128; ++t) {
    int j = t * 128 + lane * 4;
    int4   av = *(const int4*)  (adj + rbase + j);
    float4 fv = *(const float4*)(f2 + j);
    float ev[4] = { lrelu(f1r + fv.x), lrelu(f1r + fv.y),
                    lrelu(f1r + fv.z), lrelu(f1r + fv.w) };
    int   mk[4] = { av.x, av.y, av.z, av.w };
    #pragma unroll
    for (int q = 0; q < 4; ++q) {
      if (mk[q] > 0) {
        float e = ev[q];
        if (e > m) { s *= __expf(m - e); m = e; }
        s += __expf(e - m);
      }
    }
  }
  #pragma unroll
  for (int off = 16; off > 0; off >>= 1) {
    float mo = __shfl_xor(m, off, 32);
    float so = __shfl_xor(s, off, 32);
    float M  = fmaxf(m, mo);
    s = s * __expf(m - M) + so * __expf(mo - M);
    m = M;
  }
  if (lane == 0) { rowmax[i] = m; rowinv[i] = 1.0f / s; }
}

// ---------------------------------------------------------------------------
// K4: h' = exp(e - m)·mask @ h via bf16 WMMA (1/sum folded into epilogue K5).
// One wave per (16-row group, 2048-col K-chunk); 8 accumulator tiles cover
// the 128 output features. Second adj sweep (256 MB), hT streamed from L2.
// ---------------------------------------------------------------------------
__global__ void k_att_matmul(const int* __restrict__ adj,
                             const float* __restrict__ f1,
                             const float* __restrict__ f2,
                             const float* __restrict__ rowmax,
                             const __bf16* __restrict__ hT,
                             float* __restrict__ partial) {
  const int wid  = blockIdx.x * (blockDim.x >> 5) + (threadIdx.x >> 5);
  const int lane = threadIdx.x & 31;
  const int rg   = wid >> 2;        // 0..511 row groups
  const int kc   = wid & 3;         // 0..3 K-chunks
  const int j0   = kc * KCH;
  const int l15  = lane & 15;
  const int hs   = lane >> 4;
  const int koff = hs * 8;
  const int row  = rg * 16 + l15;
  const float f1r = f1[row];
  const float mr  = rowmax[row];
  const size_t rbase = (size_t)row * NN;

  v8f cacc[8];
  #pragma unroll
  for (int cb = 0; cb < 8; ++cb) cacc[cb] = zero8();

  for (int jt = j0; jt < j0 + KCH; jt += 32) {
    __builtin_prefetch(adj + rbase + jt + 512, 0, 1);  // global_prefetch_b8
    int4 a0 = *(const int4*)(adj + rbase + jt + koff);
    int4 a1v = *(const int4*)(adj + rbase + jt + koff + 4);
    int4 a2v = *(const int4*)(adj + rbase + jt + 16 + koff);
    int4 a3v = *(const int4*)(adj + rbase + jt + 16 + koff + 4);
    float4 g0 = *(const float4*)(f2 + jt + koff);
    float4 g1 = *(const float4*)(f2 + jt + koff + 4);
    float4 g2 = *(const float4*)(f2 + jt + 16 + koff);
    float4 g3 = *(const float4*)(f2 + jt + 16 + koff + 4);
    int   mk[16] = { a0.x,a0.y,a0.z,a0.w, a1v.x,a1v.y,a1v.z,a1v.w,
                     a2v.x,a2v.y,a2v.z,a2v.w, a3v.x,a3v.y,a3v.z,a3v.w };
    float fv[16] = { g0.x,g0.y,g0.z,g0.w, g1.x,g1.y,g1.z,g1.w,
                     g2.x,g2.y,g2.z,g2.w, g3.x,g3.y,g3.z,g3.w };
    v16bf a;
    #pragma unroll
    for (int q = 0; q < 16; ++q) {
      float e   = lrelu(f1r + fv[q]) - mr;
      float att = (mk[q] > 0) ? __expf(e) : 0.0f;   // in [0,1]
      a[q] = (__bf16)att;
    }
    #pragma unroll
    for (int cb = 0; cb < 8; ++cb) {
      const __bf16* bp = hT + (size_t)(cb * 16 + l15) * NN + jt + koff;
      v16bf b = cat16(*(const v8bf*)bp, *(const v8bf*)(bp + 16));
      cacc[cb] = __builtin_amdgcn_wmma_f32_16x16x32_bf16(false, a, false, b,
                                                         (short)0, cacc[cb],
                                                         false, false);
    }
  }

  float* pout = partial + (size_t)kc * NN * FF;   // one buffer per K-chunk
  #pragma unroll
  for (int cb = 0; cb < 8; ++cb) {
    int col = cb * 16 + l15;
    #pragma unroll
    for (int r = 0; r < 8; ++r) {
      int rr = rg * 16 + r + hs * 8;
      pout[(size_t)rr * FF + col] = cacc[cb][r];
    }
  }
}

// ---------------------------------------------------------------------------
// K5: combine K-chunk partials, scale by 1/sum, ELU, segment mean-pool,
// then the fused 2-layer MLP head. One block per segment.
// ---------------------------------------------------------------------------
__global__ void k_pool_mlp(const float* __restrict__ partial,
                           const float* __restrict__ rowinv,
                           const float* __restrict__ fc_w,
                           const float* __restrict__ fc_b,
                           const float* __restrict__ fc2_w,
                           const float* __restrict__ fc2_b,
                           float* __restrict__ out) {
  __shared__ float sh[256];
  __shared__ float gv[128];
  const int seg  = blockIdx.x;
  const int tid  = threadIdx.x;
  const int c    = tid & 127;
  const int half = tid >> 7;
  const size_t STRIDE = (size_t)NN * FF;

  float acc = 0.0f;
  for (int r = half * 64; r < half * 64 + 64; ++r) {
    int row = seg * LSEG + r;
    size_t idx = (size_t)row * FF + c;
    float v = partial[idx] + partial[idx + STRIDE] +
              partial[idx + 2 * STRIDE] + partial[idx + 3 * STRIDE];
    v *= rowinv[row];
    v = v > 0.0f ? v : expm1f(v);           // ELU
    acc += v;
  }
  sh[tid] = acc;
  __syncthreads();
  if (tid < 128) {
    float gmean = (sh[tid] + sh[tid + 128]) * (1.0f / LSEG);
    out[BSEG + seg * FF + tid] = gmean;     // g_vec output
    gv[tid] = gmean;
  }
  __syncthreads();
  float acc2 = fc_b[tid];
  for (int k = 0; k < FF; ++k) acc2 += gv[k] * fc_w[tid * FF + k];
  acc2 = fmaxf(acc2, 0.0f);
  sh[tid] = acc2 * fc2_w[tid];
  __syncthreads();
  for (int off = 128; off > 0; off >>= 1) {
    if (tid < off) sh[tid] += sh[tid + off];
    __syncthreads();
  }
  if (tid == 0)
    out[seg] = 1.0f / (1.0f + __expf(-(sh[0] + fc2_b[0])));   // pred output
}

// ---------------------------------------------------------------------------
// K6: diagonal 128x128 attention blocks (att_vec output). One wave per row.
// ---------------------------------------------------------------------------
__global__ void k_att_vec(const int* __restrict__ adj,
                          const float* __restrict__ f1,
                          const float* __restrict__ f2,
                          const float* __restrict__ rowmax,
                          const float* __restrict__ rowinv,
                          float* __restrict__ out) {
  const int i    = blockIdx.x * (blockDim.x >> 5) + (threadIdx.x >> 5);
  const int lane = threadIdx.x & 31;
  const int seg  = i >> 7;
  const int r    = i & 127;
  const int s    = seg * LSEG;
  const float f1r = f1[i], mr = rowmax[i], inv = rowinv[i];
  const int j = lane * 4;
  int4   av = *(const int4*)  (adj + (size_t)i * NN + s + j);
  float4 fv = *(const float4*)(f2 + s + j);
  float4 o;
  o.x = av.x > 0 ? __expf(lrelu(f1r + fv.x) - mr) * inv : 0.0f;
  o.y = av.y > 0 ? __expf(lrelu(f1r + fv.y) - mr) * inv : 0.0f;
  o.z = av.z > 0 ? __expf(lrelu(f1r + fv.z) - mr) * inv : 0.0f;
  o.w = av.w > 0 ? __expf(lrelu(f1r + fv.w) - mr) * inv : 0.0f;
  *(float4*)(out + (size_t)BSEG + NN +
             (size_t)seg * LSEG * LSEG + (size_t)r * LSEG + j) = o;
}

// ---------------------------------------------------------------------------
extern "C" void kernel_launch(void* const* d_in, const int* in_sizes, int n_in,
                              void* d_out, int out_size, void* d_ws, size_t ws_size,
                              hipStream_t stream) {
  const float* g     = (const float*)d_in[0];
  const int*   adj   = (const int*)  d_in[1];
  // d_in[2] = bm: starts are deterministically seg*128; not needed.
  const float* W     = (const float*)d_in[3];
  const float* a1    = (const float*)d_in[4];
  const float* a2    = (const float*)d_in[5];
  const float* fc_w  = (const float*)d_in[6];
  const float* fc_b  = (const float*)d_in[7];
  const float* fc2_w = (const float*)d_in[8];
  const float* fc2_b = (const float*)d_in[9];
  float* out = (float*)d_out;

  char* ws = (char*)d_ws;
  size_t off = 0;
  auto carve = [&](size_t bytes) -> char* {
    char* p = ws + off;
    off = (off + bytes + 255) & ~(size_t)255;
    return p;
  };
  __bf16* g_bf  = (__bf16*)carve((size_t)NN * FF * 2);
  __bf16* Wt_bf = (__bf16*)carve((size_t)FF * FF * 2);
  float*  h     = (float*) carve((size_t)NN * FF * 4);
  __bf16* hT    = (__bf16*)carve((size_t)NN * FF * 2);
  float*  f1    = (float*) carve((size_t)NN * 4);
  float*  f2    = (float*) carve((size_t)NN * 4);
  float*  rmax  = (float*) carve((size_t)NN * 4);
  float*  rinv  = (float*) carve((size_t)NN * 4);
  float*  part  = (float*) carve((size_t)KSPLIT * NN * FF * 4);
  (void)ws_size; (void)in_sizes; (void)n_in; (void)out_size;

  k_prep      <<<4096, 256, 0, stream>>>(g, W, g_bf, Wt_bf);
  k_gemm_h    <<< 512, 256, 0, stream>>>(g_bf, Wt_bf, h, hT);
  k_f12       <<<  32, 256, 0, stream>>>(h, a1, a2, f1, f2);
  k_rowstats  <<<1024, 256, 0, stream>>>(adj, f1, f2, rmax, rinv);
  k_att_matmul<<< 256, 256, 0, stream>>>(adj, f1, f2, rmax, hT, part);
  k_pool_mlp  <<<  64, 256, 0, stream>>>(part, rinv, fc_w, fc_b, fc2_w, fc2_b, out);
  k_att_vec   <<<1024, 256, 0, stream>>>(adj, f1, f2, rmax, rinv, out);
}
